// GraphPropagationBlock_88983132439136
// MI455X (gfx1250) — matrix operand order repaired and checked by reading, not compiled, not run on hardware
//
#include <hip/hip_runtime.h>
#include <math.h>

// ---------------- types for WMMA / async DMA ----------------
typedef __attribute__((ext_vector_type(16))) __bf16        v16bf;
typedef __attribute__((ext_vector_type(8)))  float         v8f;
typedef __attribute__((ext_vector_type(8)))  unsigned int  v8u;
typedef __attribute__((ext_vector_type(4)))  int           v4i;
typedef __attribute__((address_space(1)))    v4i           g_v4i;   // global int4
typedef __attribute__((address_space(3)))    v4i           l_v4i;   // LDS int4

// ---------------- problem constants ----------------
#define BDIM   64
#define NTOK   197
#define CDIM   768
#define HNUM   12
#define DH     64
#define RELIM  98
#define KKEEP  99
#define M1     (BDIM * NTOK)   // 12608 rows (attention path)
#define M2     (BDIM * KKEEP)  // 6336 rows (MLP path)
#define THRCNT 1941            // thr_idx = int(98*99*0.2) = 1940 -> need count(>=v) >= 1941

#ifndef __has_builtin
#define __has_builtin(x) 0
#endif

// CDNA5 async global->LDS DMA path (ASYNCcnt-tracked), with plain-load fallback
#if __has_builtin(__builtin_amdgcn_global_load_async_to_lds_b128)
#define USE_ASYNC_LDS 1
#define ASYNC_CP128(dstLds, srcGlb)                                          \
  __builtin_amdgcn_global_load_async_to_lds_b128(                            \
      (g_v4i*)(void*)(srcGlb), (l_v4i*)(void*)(dstLds), 0, 0)
#else
#define USE_ASYNC_LDS 0
#endif

__device__ __forceinline__ void wait_async_le4() {
#if __has_builtin(__builtin_amdgcn_s_wait_asynccnt)
  __builtin_amdgcn_s_wait_asynccnt(4);
#else
  asm volatile("s_wait_asynccnt 0x4" ::: "memory");
#endif
}
__device__ __forceinline__ void wait_async_0() {
#if __has_builtin(__builtin_amdgcn_s_wait_asynccnt)
  __builtin_amdgcn_s_wait_asynccnt(0);
#else
  asm volatile("s_wait_asynccnt 0x0" ::: "memory");
#endif
}

__device__ __forceinline__ unsigned short f2bf(float f) {
  unsigned int u = __float_as_uint(f);
  u += 0x7FFFu + ((u >> 16) & 1u);           // round-to-nearest-even
  return (unsigned short)(u >> 16);
}

// one 32-K step of the 64x32 wave tile: 4 A-frags x 2 B-frags -> 8 WMMAs
__device__ __forceinline__ void wmma_tile(const unsigned short* __restrict__ lAc,
                                          const unsigned short* __restrict__ lBc,
                                          int waveM, int waveN, int mIn, int hl,
                                          v8f acc[4][2])
{
  // B fragments: 16-bit B 32x16, lane n=lane%16, element e <-> k = 16*hl + e
  v16bf bF[2];
#pragma unroll
  for (int ns = 0; ns < 2; ns++) {
    const unsigned short* basep = lBc + (waveN * 32 + ns * 16 + mIn) * 40 + 16 * hl;
    v8u u;
#pragma unroll
    for (int i = 0; i < 8; i++) u[i] = *(const unsigned int*)(basep + 2 * i);
    bF[ns] = __builtin_bit_cast(v16bf, u);
  }
  // A fragments: 16-bit A 16x32, lane m=lane%16; dword i: k=16*hl+2i
#pragma unroll
  for (int ms = 0; ms < 4; ms++) {
    const unsigned short* baseA = lAc + (waveM * 64 + ms * 16 + mIn) * 40;
    v8u ua;
#pragma unroll
    for (int i = 0; i < 8; i++) {
      const int kk = 16 * hl + 2 * i;
      ua[i] = *(const unsigned int*)(baseA + kk);
    }
    const v16bf aF = __builtin_bit_cast(v16bf, ua);
#pragma unroll
    for (int ns = 0; ns < 2; ns++) {
      acc[ms][ns] = __builtin_amdgcn_wmma_f32_16x16x32_bf16(
          false, aF, false, bF[ns], (short)0, acc[ms][ns], false, false);
    }
  }
}

// ============================================================
// Generic bf16 WMMA GEMM:  out = act(A[M,K] @ Bt[N,K]^T + bias) + res
// Block tile 128x128, 8 waves (2x4), wave tile 64x32.
// Async-DMA triple-buffered LDS pipeline, 1 barrier / K-step,
// software-pipelined in groups of 3 K-steps (requires K % 96 == 0).
// ============================================================
__global__ __launch_bounds__(256)
void gemm_bf16_wmma(const unsigned short* __restrict__ A,
                    const unsigned short* __restrict__ Bt,
                    const float* __restrict__ bias,
                    const float* __restrict__ res,
                    float* __restrict__ outF,
                    unsigned short* __restrict__ outBf,
                    int M, int N, int K, int act)
{
#if USE_ASYNC_LDS
  __shared__ __align__(16) unsigned short lA[3][128 * 40];  // 3 x 10240 B
  __shared__ __align__(16) unsigned short lB[3][128 * 40];
#else
  __shared__ __align__(16) unsigned short lA[2][128 * 40];
  __shared__ __align__(16) unsigned short lB[2][128 * 40];
#endif
  const int tid   = threadIdx.x;
  const int bm    = blockIdx.y * 128;
  const int bn    = blockIdx.x * 128;
  const int lane  = tid & 31;
  const int wave  = tid >> 5;
  const int waveM = wave >> 2;   // 0..1
  const int waveN = wave & 3;    // 0..3
  const int mIn   = lane & 15;
  const int hl    = lane >> 4;

  // fixed staging assignment: this thread stages rows r0 and r0+64, 16B chunk (tid&3)
  const int r0   = tid >> 2;               // 0..63
  const int soff = (tid & 3) << 3;         // halfs: 0,8,16,24
  int ga0 = bm + r0;       if (ga0 >= M) ga0 = M - 1;   // clamp M edge
  int ga1 = bm + r0 + 64;  if (ga1 >= M) ga1 = M - 1;
  const unsigned short* pA0 = A  + (size_t)ga0 * K + soff;
  const unsigned short* pA1 = A  + (size_t)ga1 * K + soff;
  const unsigned short* pB0 = Bt + (size_t)(bn + r0) * K + soff;       // N % 128 == 0
  const unsigned short* pB1 = Bt + (size_t)(bn + r0 + 64) * K + soff;
  const int lo0 = r0 * 40 + soff;
  const int lo1 = (r0 + 64) * 40 + soff;

  v8f acc[4][2];
#pragma unroll
  for (int i = 0; i < 4; i++)
#pragma unroll
    for (int j = 0; j < 2; j++)
#pragma unroll
      for (int r = 0; r < 8; r++) acc[i][j][r] = 0.0f;

  const int nk = K >> 5;   // K / 32; nk in {24, 96}: multiple of 3, >= 6

#if USE_ASYNC_LDS
#define ISSUE_TILE(buf, t) do {                                               \
    const int _ko = (t) << 5;                                                 \
    ASYNC_CP128(&lA[buf][lo0], pA0 + _ko); ASYNC_CP128(&lA[buf][lo1], pA1 + _ko); \
    ASYNC_CP128(&lB[buf][lo0], pB0 + _ko); ASYNC_CP128(&lB[buf][lo1], pB1 + _ko); \
  } while (0)
#define PIPE_STEP(cbuf, ibuf, itile) do {                                     \
    wait_async_le4();                      /* tile for cbuf has landed */     \
    __syncthreads();                       /* visible to all waves */         \
    ISSUE_TILE(ibuf, itile);               /* DMA 2 tiles ahead */            \
    wmma_tile(lA[cbuf], lB[cbuf], waveM, waveN, mIn, hl, acc);                \
  } while (0)

  // prologue: DMA tiles 0 and 1 into buffers 0 and 1
  ISSUE_TILE(0, 0);
  ISSUE_TILE(1, 1);
  // steady state: constant buffer rotation, groups of 3 K-steps
  int kt = 0;
  for (; kt < nk - 3; kt += 3) {
    PIPE_STEP(0, 2, kt + 2);
    PIPE_STEP(1, 0, kt + 3);
    PIPE_STEP(2, 1, kt + 4);
  }
  // tail group: tiles nk-3 (buf0), nk-2 (buf1), nk-1 (buf2)
  PIPE_STEP(0, 2, nk - 1);
  wait_async_le4();
  __syncthreads();
  wmma_tile(lA[1], lB[1], waveM, waveN, mIn, hl, acc);
  wait_async_0();
  __syncthreads();
  wmma_tile(lA[2], lB[2], waveM, waveN, mIn, hl, acc);
#undef PIPE_STEP
#undef ISSUE_TILE
#else
  // fallback: register-staged double buffer with plain loads
  uint4 rA0 = *(const uint4*)pA0, rA1 = *(const uint4*)pA1;
  uint4 rB0 = *(const uint4*)pB0, rB1 = *(const uint4*)pB1;
  for (int kt = 0; kt < nk; ++kt) {
    const int cur = kt & 1;
    *(uint4*)(&lA[cur][lo0]) = rA0;  *(uint4*)(&lA[cur][lo1]) = rA1;
    *(uint4*)(&lB[cur][lo0]) = rB0;  *(uint4*)(&lB[cur][lo1]) = rB1;
    if (kt + 1 < nk) {
      const int ko = (kt + 1) << 5;
      rA0 = *(const uint4*)(pA0 + ko); rA1 = *(const uint4*)(pA1 + ko);
      rB0 = *(const uint4*)(pB0 + ko); rB1 = *(const uint4*)(pB1 + ko);
    }
    __syncthreads();
    wmma_tile(lA[cur], lB[cur], waveM, waveN, mIn, hl, acc);
    __syncthreads();
  }
#endif

  // epilogue: C/D layout — VGPR r: m = r + 8*hl, n = lane%16
#pragma unroll
  for (int ms = 0; ms < 4; ms++) {
#pragma unroll
    for (int ns = 0; ns < 2; ns++) {
      const int nCol = bn + waveN * 32 + ns * 16 + mIn;
#pragma unroll
      for (int r = 0; r < 8; r++) {
        const int m = bm + waveM * 64 + ms * 16 + r + 8 * hl;
        if (m < M) {
          float vv = acc[ms][ns][r];
          if (bias) vv += bias[nCol];
          if (act)  vv = 0.5f * vv * (1.0f + erff(vv * 0.70710678118654752f)); // exact GELU
          if (res)  vv += res[(size_t)m * N + nCol];
          const size_t o = (size_t)m * N + nCol;
          if (outBf) outBf[o] = f2bf(vv); else outF[o] = vv;
        }
      }
    }
  }
}

// ---------------- fp32 -> bf16 transpose-convert: dst[n*K+k] = src[k*N+n] ----------------
__global__ __launch_bounds__(256)
void convT(const float* __restrict__ src, unsigned short* __restrict__ dst, int Kd, int Nd)
{
  const size_t idx = (size_t)blockIdx.x * 256 + threadIdx.x;
  if (idx >= (size_t)Kd * Nd) return;
  const int n = (int)(idx % Nd);
  const int k = (int)(idx / Nd);
  dst[(size_t)n * Kd + k] = f2bf(src[idx]);
}

// ---------------- LayerNorm (row=block), fp32 in -> bf16 out ----------------
__global__ __launch_bounds__(256)
void ln_bf16(const float* __restrict__ x, const float* __restrict__ g,
             const float* __restrict__ bta, unsigned short* __restrict__ out)
{
  const int row = blockIdx.x;
  const int tid = threadIdx.x;
  __shared__ float red[256];
  const float* xr = x + (size_t)row * CDIM;
  const float v0 = xr[tid], v1 = xr[tid + 256], v2 = xr[tid + 512];
  red[tid] = v0 + v1 + v2;
  __syncthreads();
  for (int st = 128; st > 0; st >>= 1) { if (tid < st) red[tid] += red[tid + st]; __syncthreads(); }
  const float mean = red[0] * (1.0f / CDIM);
  __syncthreads();
  const float d0 = v0 - mean, d1 = v1 - mean, d2 = v2 - mean;
  red[tid] = d0 * d0 + d1 * d1 + d2 * d2;
  __syncthreads();
  for (int st = 128; st > 0; st >>= 1) { if (tid < st) red[tid] += red[tid + st]; __syncthreads(); }
  const float inv = rsqrtf(red[0] * (1.0f / CDIM) + 1e-5f);
  unsigned short* orow = out + (size_t)row * CDIM;
  orow[tid]       = f2bf(d0 * inv * g[tid]       + bta[tid]);
  orow[tid + 256] = f2bf(d1 * inv * g[tid + 256] + bta[tid + 256]);
  orow[tid + 512] = f2bf(d2 * inv * g[tid + 512] + bta[tid + 512]);
}

// ---------------- split qkv [M1,2304] -> q*scale, k, v as [B,H,N,Dh] ----------------
__global__ __launch_bounds__(256)
void split_qkv(const float* __restrict__ qkv, float* __restrict__ q,
               float* __restrict__ k, float* __restrict__ v)
{
  const size_t idx = (size_t)blockIdx.x * 256 + threadIdx.x;
  if (idx >= (size_t)M1 * 2304) return;
  const int    col = (int)(idx % 2304);
  const size_t row = idx / 2304;
  const int b = (int)(row / NTOK), n = (int)(row % NTOK);
  const int three = col / 768;
  const int h = (col % 768) / DH;
  const int d = col & 63;
  const float val = qkv[idx];
  const size_t dst = (((size_t)b * HNUM + h) * NTOK + n) * DH + d;
  if (three == 0)      q[dst] = val * 0.125f;  // Dh^-0.5
  else if (three == 1) k[dst] = val;
  else                 v[dst] = val;
}

// ---------------- attention: one (b,h) per block; stores full attn + bf16 context ----------------
__global__ __launch_bounds__(256)
void attn_kernel(const float* __restrict__ q, const float* __restrict__ k,
                 const float* __restrict__ v, float* __restrict__ attn,
                 unsigned short* __restrict__ aout)
{
  const int bh = blockIdx.x;
  const int b = bh / HNUM, h = bh % HNUM;
  const int tid = threadIdx.x;
  const float* qb = q + (size_t)bh * NTOK * DH;
  const float* kb = k + (size_t)bh * NTOK * DH;
  const float* vb = v + (size_t)bh * NTOK * DH;
  __shared__ float qrow[DH];
  __shared__ float p[NTOK];
  __shared__ float red[256];

  for (int n = 0; n < NTOK; n++) {
    if (tid < DH) qrow[tid] = qb[(size_t)n * DH + tid];
    __syncthreads();
    float s = -3.0e38f;
    if (tid < NTOK) {
      float a = 0.0f;
      const float* kr = kb + (size_t)tid * DH;
#pragma unroll 8
      for (int i = 0; i < DH; i++) a = fmaf(qrow[i], kr[i], a);
      p[tid] = a;
      s = a;
    }
    red[tid] = s; __syncthreads();
    for (int st = 128; st > 0; st >>= 1) { if (tid < st) red[tid] = fmaxf(red[tid], red[tid + st]); __syncthreads(); }
    const float mx = red[0]; __syncthreads();
    float e = 0.0f;
    if (tid < NTOK) { e = expf(p[tid] - mx); p[tid] = e; }
    red[tid] = e; __syncthreads();
    for (int st = 128; st > 0; st >>= 1) { if (tid < st) red[tid] += red[tid + st]; __syncthreads(); }
    const float inv = 1.0f / red[0]; __syncthreads();
    if (tid < NTOK) {
      const float pr = p[tid] * inv;
      p[tid] = pr;
      attn[((size_t)bh * NTOK + n) * NTOK + tid] = pr;
    }
    __syncthreads();
    // context: out[n,d] = sum_m p[m] * v[m,d]
    const int d = tid & 63, chunk = tid >> 6;
    float o = 0.0f;
    for (int m = chunk; m < NTOK; m += 4)
      o = fmaf(p[m], vb[(size_t)m * DH + d], o);
    red[tid] = o; __syncthreads();
    if (chunk == 0) {
      const float oo = red[d] + red[64 + d] + red[128 + d] + red[192 + d];
      aout[((size_t)(b * NTOK + n)) * CDIM + h * DH + d] = f2bf(oo);
    }
    __syncthreads();
  }
}

// ---------------- head-mean attention diagonal (excluding CLS) ----------------
__global__ __launch_bounds__(256)
void diag_kernel(const float* __restrict__ attn, float* __restrict__ diag)
{
  const int idx = blockIdx.x * 256 + threadIdx.x;
  if (idx >= BDIM * (NTOK - 1)) return;
  const int b = idx / (NTOK - 1), j = idx % (NTOK - 1);
  float s = 0.0f;
  for (int h = 0; h < HNUM; h++)
    s += attn[(((size_t)(b * HNUM + h)) * NTOK + (j + 1)) * NTOK + (j + 1)];
  diag[idx] = s * (1.0f / HNUM);
}

// ---------------- stable-rank token selection; outputs sorted kept/elim ----------------
__global__ __launch_bounds__(256)
void select_kernel(const float* __restrict__ diag, int* __restrict__ kept, int* __restrict__ elim)
{
  const int b = blockIdx.x;
  const int tid = threadIdx.x;
  __shared__ float dg[NTOK - 1];
  __shared__ int rk[NTOK - 1];
  if (tid < NTOK - 1) dg[tid] = diag[b * (NTOK - 1) + tid];
  __syncthreads();
  if (tid < NTOK - 1) {
    const float mine = dg[tid];
    int r = 0;
    for (int j = 0; j < NTOK - 1; j++) {
      const float o = dg[j];
      r += (o > mine) || (o == mine && j < tid);  // stable argsort(-diag)
    }
    rk[tid] = r;
  }
  __syncthreads();
  if (tid == 0) {
    int kc = 0, ec = 0;
    kept[b * KKEEP + kc++] = 0;  // CLS always kept; scan order => sorted ascending
    for (int t = 0; t < NTOK - 1; t++) {
      if (rk[t] < (NTOK - 1 - RELIM)) kept[b * KKEEP + kc++] = t + 1;
      else                            elim[b * RELIM + ec++] = t + 1;
    }
  }
}

// ---------------- gather w[K,R] and exact top-k threshold (bit-level binary search) ----------------
__global__ __launch_bounds__(256)
void gather_thresh(const float* __restrict__ attn, const int* __restrict__ kept,
                   const int* __restrict__ elim, float* __restrict__ wthr)
{
  const int bh = blockIdx.x;
  const int b = bh / HNUM;
  const int tid = threadIdx.x;
  __shared__ float w[KKEEP * RELIM];  // 9702 floats
  __shared__ int kid[KKEEP];
  __shared__ int eid[RELIM];
  __shared__ int cred[256];
  if (tid < KKEEP) kid[tid] = kept[b * KKEEP + tid];
  if (tid < RELIM) eid[tid] = elim[b * RELIM + tid];
  __syncthreads();
  const float* ab = attn + (size_t)bh * NTOK * NTOK;
  for (int i = tid; i < KKEEP * RELIM; i += 256) {
    const int kk = i / RELIM, ee = i % RELIM;
    w[i] = ab[(size_t)kid[kk] * NTOK + eid[ee]];
  }
  __syncthreads();
  // exact k-th largest: max u with count(x >= float(u)) >= THRCNT (softmax values >= 0)
  unsigned lo = 0u, hi = 0x7f800000u;
  while (lo < hi) {
    const unsigned mid = lo + (hi - lo + 1u) / 2u;
    const float t = __uint_as_float(mid);
    int c = 0;
    for (int i = tid; i < KKEEP * RELIM; i += 256) c += (w[i] >= t) ? 1 : 0;
    cred[tid] = c; __syncthreads();
    for (int st = 128; st > 0; st >>= 1) { if (tid < st) cred[tid] += cred[tid + st]; __syncthreads(); }
    const int total = cred[0]; __syncthreads();
    if (total >= THRCNT) lo = mid; else hi = mid - 1u;
  }
  const float thr = __uint_as_float(lo);
  float* wb = wthr + (size_t)bh * KKEEP * RELIM;
  for (int i = tid; i < KKEEP * RELIM; i += 256) {
    const float vv = w[i];
    wb[i] = (vv >= thr) ? vv : 0.0f;
  }
}

// ---------------- x2[b,k,:] = x1[b,kept[k],:] + alpha * sum_e w[b,h,k,e] x1[b,elim[e],:] ----------------
__global__ __launch_bounds__(256)
void propagate(const float* __restrict__ x1, const float* __restrict__ wthr,
               const int* __restrict__ kept, const int* __restrict__ elim,
               float* __restrict__ x2)
{
  const int bk = blockIdx.x;               // b*KKEEP + kk
  const int b = bk / KKEEP, kk = bk % KKEEP;
  const int tid = threadIdx.x;
  __shared__ float wl[HNUM * RELIM];
  __shared__ int eid[RELIM];
  if (tid < RELIM) eid[tid] = elim[b * RELIM + tid];
  for (int i = tid; i < HNUM * RELIM; i += 256) {
    const int h = i / RELIM, e = i % RELIM;
    wl[i] = wthr[(((size_t)(b * HNUM + h)) * KKEEP + kk) * RELIM + e];
  }
  __syncthreads();
  const int kidx = kept[b * KKEEP + kk];
  const float* xb = x1 + (size_t)b * NTOK * CDIM;
  for (int c = tid; c < CDIM; c += 256) {
    const int h = c >> 6;
    float a = 0.0f;
    for (int e = 0; e < RELIM; e++)
      a = fmaf(wl[h * RELIM + e], xb[(size_t)eid[e] * CDIM + c], a);
    x2[(size_t)bk * CDIM + c] = xb[(size_t)kidx * CDIM + c] + 0.1f * a;
  }
}

// ============================================================
extern "C" void kernel_launch(void* const* d_in, const int* in_sizes, int n_in,
                              void* d_out, int out_size, void* d_ws, size_t ws_size,
                              hipStream_t stream)
{
  (void)in_sizes; (void)n_in; (void)out_size; (void)ws_size;
  const float* x      = (const float*)d_in[0];
  const float* n1g    = (const float*)d_in[1];
  const float* n1b    = (const float*)d_in[2];
  const float* qkv_w  = (const float*)d_in[3];
  const float* proj_w = (const float*)d_in[4];
  const float* proj_b = (const float*)d_in[5];
  const float* n2g    = (const float*)d_in[6];
  const float* n2b    = (const float*)d_in[7];
  const float* fc1_w  = (const float*)d_in[8];
  const float* fc1_b  = (const float*)d_in[9];
  const float* fc2_w  = (const float*)d_in[10];
  const float* fc2_b  = (const float*)d_in[11];

  char* base = (char*)d_ws;
  size_t off = 0;
  auto alloc = [&](size_t bytes) -> char* {
    char* p = base + off;
    off += (bytes + 255) & ~(size_t)255;
    return p;
  };

  // bf16 transposed weights
  unsigned short* qkvwT  = (unsigned short*)alloc((size_t)2304 * 768 * 2);
  unsigned short* projwT = (unsigned short*)alloc((size_t)768  * 768 * 2);
  unsigned short* fc1wT  = (unsigned short*)alloc((size_t)3072 * 768 * 2);
  unsigned short* fc2wT  = (unsigned short*)alloc((size_t)768  * 3072 * 2);

  // region1: qkv fp32 (dead after split) overlaid with full attn matrix
  const size_t qkvBytes  = (size_t)M1 * 2304 * 4;
  const size_t attnBytes = (size_t)BDIM * HNUM * NTOK * NTOK * 4;
  char* region1 = alloc(qkvBytes > attnBytes ? qkvBytes : attnBytes);
  float* qkvf = (float*)region1;
  float* attn = (float*)region1;

  // region2: q/k/v (dead after attention) overlaid with wthr/x2/h2/g1
  const size_t headBytes = (size_t)M1 * CDIM * 4;  // per q/k/v
  char* region2 = alloc(3 * headBytes);
  float* qh = (float*)region2;
  float* kh = (float*)(region2 + headBytes);
  float* vh = (float*)(region2 + 2 * headBytes);
  size_t o2 = 0;
  auto sub2 = [&](size_t bytes) -> char* {
    char* p = region2 + o2;
    o2 += (bytes + 255) & ~(size_t)255;
    return p;
  };
  float*          wthr = (float*)sub2((size_t)BDIM * HNUM * KKEEP * RELIM * 4);
  float*          x2   = (float*)sub2((size_t)M2 * CDIM * 4);
  unsigned short* h2bf = (unsigned short*)sub2((size_t)M2 * CDIM * 2);
  unsigned short* g1bf = (unsigned short*)sub2((size_t)M2 * 3072 * 2);

  // region3: ln1 output (dead after QKV GEMM) overlaid with bf16 attention context
  char* region3 = alloc((size_t)M1 * CDIM * 2);
  unsigned short* h1bf   = (unsigned short*)region3;
  unsigned short* aoutbf = (unsigned short*)region3;

  float* x1   = (float*)alloc((size_t)M1 * CDIM * 4);
  float* diag = (float*)alloc((size_t)BDIM * (NTOK - 1) * 4);
  int*   kept = (int*)alloc((size_t)BDIM * KKEEP * 4);
  int*   elim = (int*)alloc((size_t)BDIM * RELIM * 4);

  // 1) weights -> bf16, transposed
  convT<<<(768 * 2304 + 255) / 256, 256, 0, stream>>>(qkv_w,  qkvwT, 768,  2304);
  convT<<<(768 * 768  + 255) / 256, 256, 0, stream>>>(proj_w, projwT, 768, 768);
  convT<<<(768 * 3072 + 255) / 256, 256, 0, stream>>>(fc1_w,  fc1wT, 768,  3072);
  convT<<<(3072 * 768 + 255) / 256, 256, 0, stream>>>(fc2_w,  fc2wT, 3072, 768);

  // 2) LN1 -> bf16
  ln_bf16<<<M1, 256, 0, stream>>>(x, n1g, n1b, h1bf);

  // 3) QKV GEMM: [12608,768] x [768,2304]
  gemm_bf16_wmma<<<dim3(2304 / 128, (M1 + 127) / 128), 256, 0, stream>>>(
      h1bf, qkvwT, nullptr, nullptr, qkvf, nullptr, M1, 2304, 768, 0);

  // 4) split + scale q
  split_qkv<<<(unsigned)(((size_t)M1 * 2304 + 255) / 256), 256, 0, stream>>>(qkvf, qh, kh, vh);

  // 5) attention (stores attn fp32 + bf16 context)
  attn_kernel<<<BDIM * HNUM, 256, 0, stream>>>(qh, kh, vh, attn, aoutbf);

  // 6/7) diag + selection
  diag_kernel<<<(BDIM * (NTOK - 1) + 255) / 256, 256, 0, stream>>>(attn, diag);
  select_kernel<<<BDIM, 256, 0, stream>>>(diag, kept, elim);

  // 8) gather + threshold
  gather_thresh<<<BDIM * HNUM, 256, 0, stream>>>(attn, kept, elim, wthr);

  // 9) proj GEMM with bias + residual(x): x1 = x + ctx @ proj_w + proj_b
  gemm_bf16_wmma<<<dim3(768 / 128, (M1 + 127) / 128), 256, 0, stream>>>(
      aoutbf, projwT, proj_b, x, x1, nullptr, M1, 768, 768, 0);

  // 10) sparse propagation -> x2 [B,K,C]
  propagate<<<M2, 256, 0, stream>>>(x1, wthr, kept, elim, x2);

  // 11) LN2 -> bf16
  ln_bf16<<<M2, 256, 0, stream>>>(x2, n2g, n2b, h2bf);

  // 12) fc1 GEMM + bias + exact GELU, bf16 out
  gemm_bf16_wmma<<<dim3(3072 / 128, (M2 + 127) / 128), 256, 0, stream>>>(
      h2bf, fc1wT, fc1_b, nullptr, nullptr, g1bf, M2, 3072, 768, 1);

  // 13) fc2 GEMM + bias + residual(x2) -> d_out fp32 [B,K,C]
  gemm_bf16_wmma<<<dim3(768 / 128, (M2 + 127) / 128), 256, 0, stream>>>(
      g1bf, fc2wT, fc2_b, x2, (float*)d_out, nullptr, M2, 768, 3072, 0);
}
